// CCD_SurvPath_4286377362089
// MI455X (gfx1250) — matrix operand order repaired and checked by reading, not compiled, not run on hardware
//
#include <hip/hip_runtime.h>
#include <hip/hip_bf16.h>

typedef __attribute__((ext_vector_type(16))) _Float16 v16h;
typedef __attribute__((ext_vector_type(8)))  float    v8f;

#define SORT_N 8192   // next pow2 >= G=4999
#define SPAD   64     // padded P stride for S matrix

// ---------------------------------------------------------------------------
// Kernel 0: pack pathway membership into 64-bit masks; compute sizes & 1/denom
// ---------------------------------------------------------------------------
__global__ __launch_bounds__(256) void pack_kernel(const float* __restrict__ path,
                                                   unsigned long long* __restrict__ pm,
                                                   float* __restrict__ sizeArr,
                                                   float* __restrict__ invDenom,
                                                   int G, int P) {
    for (int g = blockIdx.x * 256 + threadIdx.x; g < G; g += gridDim.x * 256) {
        unsigned long long m = 0ull;
        for (int p = 0; p < P; ++p)
            if (path[(size_t)p * G + g] > 0.0f) m |= (1ull << p);
        pm[g] = m;
    }
    if (blockIdx.x == 0 && (int)threadIdx.x < P) {
        int p = threadIdx.x;
        float s = 0.0f;
        for (int g = 0; g < G; ++g) s += (path[(size_t)p * G + g] > 0.0f) ? 1.0f : 0.0f;
        sizeArr[p] = s;
        float d = (float)G - s;
        if (d < 1.0f) d = 1.0f;
        invDenom[p] = 1.0f / d;
    }
}

// ---------------------------------------------------------------------------
// Kernel 1: S[b,p] = sum_g |raw[b,g]|^0.25 * path[p,g]  via WMMA (split f16)
// One wave computes a 16x16 tile. A split as hi(f16)+residual(f16) so the
// f32-accumulated result is accurate to ~1e-6.
// ---------------------------------------------------------------------------
__global__ __launch_bounds__(32) void gemmS_kernel(const float* __restrict__ raw,
                                                   const float* __restrict__ path,
                                                   float* __restrict__ Smat,
                                                   int B, int P, int G) {
    const int lane = threadIdx.x;       // 0..31
    const int half = lane >> 4;         // 0 or 1
    const int mn   = lane & 15;
    const int brow = blockIdx.x * 16 + mn;   // A row (sample b), B=64 -> always valid
    const int pcol = blockIdx.y * 16 + mn;   // B col (pathway p), may exceed P

    v8f acc = {};
    for (int k0 = 0; k0 < G; k0 += 32) {
        v16h ah, ar, bb;
        #pragma unroll
        for (int v = 0; v < 8; ++v) {
            #pragma unroll
            for (int pos = 0; pos < 2; ++pos) {
                const int e  = v * 2 + pos;
                // A-matrix 16x32 f16 layout: VGPR v<4 -> K=half*8+(v)*2+pos,
                // v>=4 -> K=16+half*8+(v-4)*2+pos  (lane holds row M=mn)
                const int ka = ((v >= 4) ? 16 : 0) + half * 8 + (v & 3) * 2 + pos;
                const int ga = k0 + ka;
                float w = 0.0f;
                if (ga < G) {
                    float x = raw[(size_t)brow * G + ga];
                    w = sqrtf(sqrtf(fabsf(x)));     // |x|^0.25
                }
                _Float16 h = (_Float16)w;
                ah[e] = h;
                ar[e] = (_Float16)(w - (float)h);   // residual
                // B-matrix 32x16 f16 layout: same K order as A (v<4 -> half*8+v*2+pos, v>=4 -> 16+half*8+(v-4)*2+pos)
                const int kb = ((v >= 4) ? 16 : 0) + half * 8 + (v & 3) * 2 + pos;
                const int gb = k0 + kb;
                float pv = 0.0f;
                if (gb < G && pcol < P) pv = path[(size_t)pcol * G + gb];
                bb[e] = (_Float16)pv;               // 0/1 exact in f16
            }
        }
        acc = __builtin_amdgcn_wmma_f32_16x16x32_f16(false, ah, false, bb,
                                                     (short)0, acc, false, false);
        acc = __builtin_amdgcn_wmma_f32_16x16x32_f16(false, ar, false, bb,
                                                     (short)0, acc, false, false);
    }
    // D 16x16 f32: lane holds D[M = r + 8*half][N = mn] in acc[r]
    #pragma unroll
    for (int r = 0; r < 8; ++r) {
        int ob = blockIdx.x * 16 + r + 8 * half;
        int op = blockIdx.y * 16 + mn;
        Smat[(size_t)ob * SPAD + op] = acc[r];
    }
}

// ---------------------------------------------------------------------------
// Kernel 2: per-sample stable descending sort (bitonic, 64KB LDS) + gather of
// sorted weights |x|^0.25 and packed pathway hit-bits.
// key = (~orderable(x)) << 32 | gene_idx  -> ascending sort == argsort(-x)
// ---------------------------------------------------------------------------
__global__ __launch_bounds__(256) void sort_kernel(const float* __restrict__ raw,
                                                   const unsigned long long* __restrict__ pm,
                                                   float* __restrict__ wabsS,
                                                   unsigned long long* __restrict__ bits,
                                                   int G) {
    __shared__ unsigned long long keys[SORT_N];
    const int b = blockIdx.x;
    for (int i = threadIdx.x; i < SORT_N; i += 256) {
        unsigned long long key;
        if (i < G) {
            unsigned int u = __float_as_uint(raw[(size_t)b * G + i]);
            unsigned int s = u ^ ((u >> 31) ? 0xFFFFFFFFu : 0x80000000u); // ascending map
            unsigned int d = ~s;                                         // descending map
            key = ((unsigned long long)d << 32) | (unsigned int)i;
        } else {
            key = ~0ull;   // pad to the end
        }
        keys[i] = key;
    }
    __syncthreads();

    for (int k = 2; k <= SORT_N; k <<= 1) {
        for (int j = k >> 1; j > 0; j >>= 1) {
            for (int i = threadIdx.x; i < SORT_N; i += 256) {
                int ixj = i ^ j;
                if (ixj > i) {
                    unsigned long long a = keys[i], c = keys[ixj];
                    bool up = ((i & k) == 0);
                    if ((a > c) == up) { keys[i] = c; keys[ixj] = a; }
                }
            }
            __syncthreads();
        }
    }

    for (int i = threadIdx.x; i < G; i += 256) {
        unsigned int idx = (unsigned int)(keys[i] & 0xFFFFFFFFu);
        float x = raw[(size_t)b * G + idx];
        wabsS[(size_t)b * G + i] = sqrtf(sqrtf(fabsf(x)));
        bits[(size_t)b * G + i]  = pm[idx];
    }
}

// ---------------------------------------------------------------------------
// Kernel 3: one wave32 per (b,p). Wave-parallel prefix scan of weighted hits
// and hit counts; running = A/S - ((i+1)-C)/denom; shuffle arg-max of |running|
// with first-index tie-break; signed value at that position.
// ---------------------------------------------------------------------------
__global__ __launch_bounds__(32) void scan_kernel(const float* __restrict__ wabsS,
                                                  const unsigned long long* __restrict__ bits,
                                                  const float* __restrict__ Smat,
                                                  const float* __restrict__ sizeArr,
                                                  const float* __restrict__ invDenom,
                                                  float* __restrict__ out,
                                                  int G, int P) {
    const int p = blockIdx.x;
    const int b = blockIdx.y;
    const int lane = threadIdx.x;

    const float Sv   = Smat[(size_t)b * SPAD + p];
    const float invS = (Sv > 0.0f) ? (1.0f / Sv) : 1.0f;  // s<=0 -> leave w unnormalized
    const float invD = invDenom[p];

    float carryA = 0.0f, carryC = 0.0f;
    float bestAbs = -1.0f, bestVal = 0.0f;
    int   bestIdx = 0x7fffffff;

    for (int base = 0; base < G; base += 32) {
        const int i = base + lane;
        const bool valid = (i < G);
        float w = 0.0f; int hit = 0;
        if (valid) {
            w   = wabsS[(size_t)b * G + i];
            hit = (int)((bits[(size_t)b * G + i] >> p) & 1ull);
        }
        if (base + 64 < G) {   // speculative prefetch of the next chunk
            __builtin_prefetch(&wabsS[(size_t)b * G + base + 32 + lane], 0, 1);
            __builtin_prefetch(&bits[(size_t)b * G + base + 32 + lane], 0, 1);
        }
        float a = hit ? w : 0.0f;     // weighted-hit contribution
        float c = hit ? 1.0f : 0.0f;  // hit-count contribution
        // inclusive wave prefix sums
        #pragma unroll
        for (int d = 1; d < 32; d <<= 1) {
            float ta = __shfl_up(a, d, 32);
            float tc = __shfl_up(c, d, 32);
            if (lane >= d) { a += ta; c += tc; }
        }
        const float A = carryA + a;
        const float C = carryC + c;
        const float run = A * invS - ((float)(i + 1) - C) * invD;

        float av = valid ? fabsf(run) : -2.0f;
        float vv = run;
        int   ii = i;
        #pragma unroll
        for (int d = 16; d > 0; d >>= 1) {
            float oa = __shfl_down(av, d, 32);
            float ov = __shfl_down(vv, d, 32);
            int   oi = __shfl_down(ii, d, 32);
            if (oa > av || (oa == av && oi < ii)) { av = oa; vv = ov; ii = oi; }
        }
        av = __shfl(av, 0, 32); vv = __shfl(vv, 0, 32); ii = __shfl(ii, 0, 32);
        if (av > bestAbs || (av == bestAbs && ii < bestIdx)) {
            bestAbs = av; bestVal = vv; bestIdx = ii;
        }
        carryA += __shfl(a, 31, 32);
        carryC += __shfl(c, 31, 32);
    }

    if (lane == 0)
        out[(size_t)b * P + p] = (sizeArr[p] > 0.0f) ? bestVal : 0.0f;
}

// ---------------------------------------------------------------------------
extern "C" void kernel_launch(void* const* d_in, const int* in_sizes, int n_in,
                              void* d_out, int out_size, void* d_ws, size_t ws_size,
                              hipStream_t stream) {
    (void)n_in; (void)out_size; (void)ws_size;
    const float* raw  = (const float*)d_in[0];   // [B, G]
    const float* path = (const float*)d_in[1];   // [P, G]

    const int B = 64;
    const int G = in_sizes[0] / B;               // 4999
    const int P = in_sizes[1] / G;               // 50
    float* out = (float*)d_out;                  // [B, P]

    // workspace carve-up (8-byte aligned first)
    char* ws = (char*)d_ws;
    size_t off = 0;
    auto carve = [&](size_t bytes, size_t align) -> char* {
        off = (off + align - 1) & ~(align - 1);
        char* ptr = ws + off;
        off += bytes;
        return ptr;
    };
    unsigned long long* bits = (unsigned long long*)carve((size_t)B * G * 8, 8);
    unsigned long long* pm   = (unsigned long long*)carve((size_t)G * 8, 8);
    float* wabsS    = (float*)carve((size_t)B * G * 4, 4);
    float* Smat     = (float*)carve((size_t)B * SPAD * 4, 4);
    float* sizeArr  = (float*)carve((size_t)P * 4, 4);
    float* invDenom = (float*)carve((size_t)P * 4, 4);

    pack_kernel<<<dim3(32), dim3(256), 0, stream>>>(path, pm, sizeArr, invDenom, G, P);
    gemmS_kernel<<<dim3(B / 16, (P + 15) / 16), dim3(32), 0, stream>>>(raw, path, Smat, B, P, G);
    sort_kernel<<<dim3(B), dim3(256), 0, stream>>>(raw, pm, wabsS, bits, G);
    scan_kernel<<<dim3(P, B), dim3(32), 0, stream>>>(wabsS, bits, Smat, sizeArr, invDenom,
                                                     out, G, P);
}
